// HabanaAttentionImpl_76424648065205
// MI455X (gfx1250) — matrix operand order repaired and checked by reading, not compiled, not run on hardware
//
#include <hip/hip_runtime.h>
#include <hip/hip_bf16.h>

#define NBLK 128
#define BSIZE 128
#define BB 2
#define SS 2048
#define HH 32
#define HKVN 8
#define DD 128
#define GG 4
#define QK_SCALE 0.08838834764831845f
#define NEG_INF_F (-1e9f)

typedef __attribute__((ext_vector_type(16))) _Float16 v16h;
typedef __attribute__((ext_vector_type(8)))  _Float16 v8h;
typedef __attribute__((ext_vector_type(4)))  _Float16 v4h;
typedef __attribute__((ext_vector_type(8)))  float    v8f;

// 16 contiguous halves starting at p (16B aligned)
__device__ __forceinline__ v16h ld_contig16(const _Float16* p) {
  v8h lo = *(const v8h*)p;
  v8h hi = *(const v8h*)(p + 8);
  v16h r;
#pragma unroll
  for (int j = 0; j < 8; ++j) { r[j] = lo[j]; r[j + 8] = hi[j]; }
  return r;
}
// two 8-half runs at p and p+16 (WMMA A-fragment layout)
__device__ __forceinline__ v16h ld_split16(const _Float16* p) {
  v8h lo = *(const v8h*)p;
  v8h hi = *(const v8h*)(p + 16);
  v16h r;
#pragma unroll
  for (int j = 0; j < 8; ++j) { r[j] = lo[j]; r[j + 8] = hi[j]; }
  return r;
}

// ---------------- KV cache: bulk copy then token scatter ----------------

__global__ void cache_copy_kernel(const float4* __restrict__ src,
                                  float4* __restrict__ dst, int n4) {
  int i = blockIdx.x * blockDim.x + threadIdx.x;
  int stride = gridDim.x * blockDim.x;
  for (; i < n4; i += stride) dst[i] = src[i];
}

__global__ void cache_scatter_kernel(const float* __restrict__ key,
                                     const float* __restrict__ val,
                                     const int* __restrict__ bidx,
                                     const int* __restrict__ boff,
                                     float* __restrict__ kc,
                                     float* __restrict__ vc) {
  const int tok = blockIdx.x;  // 0 .. B*S-1
  const int t = threadIdx.x;   // 256 threads, one float4 each
  const size_t src = (size_t)tok * (HKVN * DD) + (size_t)t * 4;
  const int bi = bidx[tok];
  const int bo = boff[tok];
  const size_t dst = ((size_t)bi * BSIZE + bo) * (HKVN * DD) + (size_t)t * 4;
  *(float4*)(kc + dst) = *(const float4*)(key + src);
  *(float4*)(vc + dst) = *(const float4*)(val + src);
}

// ---------------- Flash-attention prefill (GQA, causal) ----------------
// 256 threads = 8 waves; each wave: 16 query rows; workgroup: 128 rows.
// Double-buffered 64-key K/V tiles in LDS; one barrier per k-block.

// Stage one 64x128 fp32 K tile and V tile into per-thread f16 registers.
// 256 threads x 8 float4 each per tensor.
__device__ __forceinline__ void load_tile_regs(const float* __restrict__ kg,
                                               const float* __restrict__ vg,
                                               int tid, v4h* kreg, v4h* vreg) {
  const size_t kd = (size_t)HKVN * DD;
#pragma unroll
  for (int i = 0; i < 8; ++i) {
    int idx = tid + i * 256;
    int r = idx >> 5, c4 = idx & 31;
    float4 k4 = *(const float4*)(kg + (size_t)r * kd + c4 * 4);
    v4h kh;
    kh[0] = (_Float16)k4.x; kh[1] = (_Float16)k4.y;
    kh[2] = (_Float16)k4.z; kh[3] = (_Float16)k4.w;
    kreg[i] = kh;
    float4 w4 = *(const float4*)(vg + (size_t)r * kd + c4 * 4);
    v4h vh;
    vh[0] = (_Float16)w4.x; vh[1] = (_Float16)w4.y;
    vh[2] = (_Float16)w4.z; vh[3] = (_Float16)w4.w;
    vreg[i] = vh;
  }
}

// Write staged tile registers into LDS (K row-major, V transposed).
__device__ __forceinline__ void store_tile_lds(_Float16 (&Kt)[64][136],
                                               _Float16 (&VtT)[128][72],
                                               int tid, const v4h* kreg,
                                               const v4h* vreg) {
#pragma unroll
  for (int i = 0; i < 8; ++i) {
    int idx = tid + i * 256;
    int r = idx >> 5, c4 = idx & 31;
    *(v4h*)&Kt[r][c4 * 4] = kreg[i];  // ds_store_b64
    VtT[c4 * 4 + 0][r] = vreg[i][0];
    VtT[c4 * 4 + 1][r] = vreg[i][1];
    VtT[c4 * 4 + 2][r] = vreg[i][2];
    VtT[c4 * 4 + 3][r] = vreg[i][3];
  }
}

// One 64-key block: S = Q K^T, online softmax, O += P V.
// MASKED: causal mask with key offset kofs relative to tile row 0.
template <bool MASKED>
__device__ __forceinline__ void attn_compute_block(
    const _Float16 (&Kt)[64][136], const _Float16 (&VtT)[128][72],
    _Float16 (&Ptw)[16][72], const v16h (&Aq)[4], v8f (&acc)[8],
    float (&mrow)[8], float (&lrow)[8], int lh, int hi, int qrel, int kofs) {
  v8f zero = {};
  float sc[4][8];
  // ---- S = Q K^T : 4 tiles of 16x16, K-dim 128 ----
#pragma unroll
  for (int t = 0; t < 4; ++t) {
    v16h bk[4];
#pragma unroll
    for (int f = 0; f < 4; ++f)
      bk[f] = ld_split16(&Kt[t * 16 + lh][f * 32 + 8 * hi]);
    v8f c = zero;
#pragma unroll
    for (int f = 0; f < 4; ++f)
      c = __builtin_amdgcn_wmma_f32_16x16x32_f16(false, Aq[f], false, bk[f],
                                                 (short)0, c, false, false);
#pragma unroll
    for (int r = 0; r < 8; ++r) {
      float s = c[r] * QK_SCALE;
      if (MASKED) {
        if (kofs + t * 16 + lh > qrel + r + 8 * hi) s += NEG_INF_F;
      }
      sc[t][r] = s;
    }
  }

  // ---- online softmax (rows live in 16-lane half-groups) ----
  float rscale[8];
#pragma unroll
  for (int r = 0; r < 8; ++r) {
    float mx = sc[0][r];
#pragma unroll
    for (int t = 1; t < 4; ++t) mx = fmaxf(mx, sc[t][r]);
#pragma unroll
    for (int d = 8; d >= 1; d >>= 1) mx = fmaxf(mx, __shfl_xor(mx, d, 32));
    float mnew = fmaxf(mrow[r], mx);
    rscale[r] = __expf(mrow[r] - mnew);
    float sum = 0.0f;
#pragma unroll
    for (int t = 0; t < 4; ++t) {
      float p = __expf(sc[t][r] - mnew);
      sc[t][r] = p;
      sum += p;
    }
#pragma unroll
    for (int d = 8; d >= 1; d >>= 1) sum += __shfl_xor(sum, d, 32);
    lrow[r] = lrow[r] * rscale[r] + sum;
    mrow[r] = mnew;
  }
#pragma unroll
  for (int t = 0; t < 8; ++t)
#pragma unroll
    for (int r = 0; r < 8; ++r) acc[t][r] *= rscale[r];

  // ---- P -> LDS (per-wave region; LDS is in-order within a wave) ----
#pragma unroll
  for (int t = 0; t < 4; ++t)
#pragma unroll
    for (int r = 0; r < 8; ++r)
      Ptw[r + 8 * hi][t * 16 + lh] = (_Float16)sc[t][r];
  __builtin_amdgcn_wave_barrier();  // scheduling fence (no hw cost)

  // ---- O += P V : 8 d-tiles, K-dim 64 ----
#pragma unroll
  for (int ks = 0; ks < 2; ++ks) {
    v16h ap = ld_split16(&Ptw[lh][ks * 32 + 8 * hi]);
#pragma unroll
    for (int td = 0; td < 8; ++td) {
      v16h bv = ld_split16(&VtT[td * 16 + lh][ks * 32 + 8 * hi]);
      acc[td] = __builtin_amdgcn_wmma_f32_16x16x32_f16(
          false, ap, false, bv, (short)0, acc[td], false, false);
    }
  }
}

__global__ __launch_bounds__(256) void attn_prefill_kernel(
    const float* __restrict__ q, const float* __restrict__ k,
    const float* __restrict__ v, float* __restrict__ out) {
  const int qt = blockIdx.x;  // 128-row query tile
  const int h  = blockIdx.y;
  const int b  = blockIdx.z;
  const int hk = h / GG;
  const int q0 = qt * 128;

  const int tid  = threadIdx.x;
  const int wave = tid >> 5;   // 0..7
  const int lane = tid & 31;
  const int lh   = lane & 15;
  const int hi   = lane >> 4;

  __shared__ __align__(32) _Float16 KtB[2][64][136];  // K tiles, double buffer
  __shared__ __align__(32) _Float16 VtB[2][128][72];  // V^T tiles, double buffer
  __shared__ __align__(32) _Float16 Pt[8][16][72];    // per-wave probs (padded)

  const size_t qd = (size_t)HH * DD;    // 4096
  const size_t kd = (size_t)HKVN * DD;  // 1024

  // ---- stage Q tile (128x128 fp32 -> f16) through both K buffers ----
  {
    const float* qg = q + ((size_t)b * SS + q0) * qd + (size_t)h * DD;
#pragma unroll
    for (int i = 0; i < 16; ++i) {
      int idx = tid + i * 256;  // 4096 float4 slots
      int r = idx >> 5, c4 = idx & 31;
      float4 val = *(const float4*)(qg + (size_t)r * qd + c4 * 4);
      v4h hq;
      hq[0] = (_Float16)val.x; hq[1] = (_Float16)val.y;
      hq[2] = (_Float16)val.z; hq[3] = (_Float16)val.w;
      *(v4h*)&KtB[r >> 6][r & 63][c4 * 4] = hq;
    }
  }
  __syncthreads();

  v16h Aq[4];
  {
    const int m = (wave & 3) * 16 + lh;  // row within the staging buffer
#pragma unroll
    for (int f = 0; f < 4; ++f)
      Aq[f] = ld_split16(&KtB[wave >> 2][m][f * 32 + 8 * hi]);
  }
  __syncthreads();  // Q fragment reads done before buffers are reloaded

  v8f zero = {};
  v8f acc[8];
#pragma unroll
  for (int t = 0; t < 8; ++t) acc[t] = zero;
  float mrow[8], lrow[8];
#pragma unroll
  for (int r = 0; r < 8; ++r) { mrow[r] = -3.0e38f; lrow[r] = 0.0f; }

  const float* kg0 = k + (size_t)b * SS * kd + (size_t)hk * DD;
  const float* vg0 = v + (size_t)b * SS * kd + (size_t)hk * DD;

  const int nfull = 2 * qt;  // unmasked 64-key blocks; then 2 diagonal blocks

  // ---- prologue: block 0 -> buffer 0 ----
  {
    v4h kreg[8], vreg[8];
    load_tile_regs(kg0, vg0, tid, kreg, vreg);
    store_tile_lds(KtB[0], VtB[0], tid, kreg, vreg);
  }
  __syncthreads();

  // ---- steady state: unmasked blocks, double buffered ----
  for (int kb = 0; kb < nfull; ++kb) {
    const int cur = kb & 1;
    v4h kreg[8], vreg[8];  // next tile's loads overlap the WMMA work below
    load_tile_regs(kg0 + (size_t)((kb + 1) * 64) * kd,
                   vg0 + (size_t)((kb + 1) * 64) * kd, tid, kreg, vreg);
    if (kb + 2 <= nfull + 1) {  // keep L2 warm two tiles ahead
      __builtin_prefetch(kg0 + (size_t)((kb + 2) * 64) * kd + tid * 8, 0, 1);
      __builtin_prefetch(vg0 + (size_t)((kb + 2) * 64) * kd + tid * 8, 0, 1);
    }
    attn_compute_block<false>(KtB[cur], VtB[cur], Pt[wave], Aq, acc, mrow,
                              lrow, lh, hi, wave * 16, 0);
    store_tile_lds(KtB[cur ^ 1], VtB[cur ^ 1], tid, kreg, vreg);
    __syncthreads();
  }

  // ---- peeled diagonal block 1 (keys q0..q0+63), load block 2 meanwhile ----
  {
    const int cur = nfull & 1;  // == 0
    v4h kreg[8], vreg[8];
    load_tile_regs(kg0 + (size_t)((nfull + 1) * 64) * kd,
                   vg0 + (size_t)((nfull + 1) * 64) * kd, tid, kreg, vreg);
    attn_compute_block<true>(KtB[cur], VtB[cur], Pt[wave], Aq, acc, mrow,
                             lrow, lh, hi, wave * 16, 0);
    store_tile_lds(KtB[cur ^ 1], VtB[cur ^ 1], tid, kreg, vreg);
    __syncthreads();
  }

  // ---- peeled diagonal block 2 (keys q0+64..q0+127) ----
  // Waves 0..3 (rows q0..q0+63) are fully masked here: skip compute.
  if (wave >= 4) {
    const int cur = (nfull + 1) & 1;  // == 1
    attn_compute_block<true>(KtB[cur], VtB[cur], Pt[wave], Aq, acc, mrow,
                             lrow, lh, hi, wave * 16, 64);
  }

  // ---- normalize + store output (fp32) ----
#pragma unroll
  for (int r = 0; r < 8; ++r) {
    float inv = 1.0f / lrow[r];
    int qrow = q0 + wave * 16 + r + 8 * hi;
    float* op = out + ((size_t)b * SS + qrow) * qd + (size_t)h * DD;
#pragma unroll
    for (int td = 0; td < 8; ++td)
      op[td * 16 + lh] = acc[td][r] * inv;
  }
}

// ---------------- launch ----------------

extern "C" void kernel_launch(void* const* d_in, const int* in_sizes, int n_in,
                              void* d_out, int out_size, void* d_ws, size_t ws_size,
                              hipStream_t stream) {
  const float* q   = (const float*)d_in[0];
  const float* k   = (const float*)d_in[1];
  const float* v   = (const float*)d_in[2];
  const float* kvc = (const float*)d_in[3];
  const int* bidx  = (const int*)d_in[4];
  const int* boff  = (const int*)d_in[5];
  // d_in[6] attn_bias: pure causal mask, computed analytically in-kernel.

  float* out_attn = (float*)d_out;
  const size_t out_elems   = (size_t)BB * SS * HH * DD;         // 16,777,216
  const size_t cache_elems = (size_t)NBLK * BSIZE * HKVN * DD;  // 16,777,216
  float* kc = out_attn + out_elems;
  float* vc = kc + cache_elems;

  // 1) copy kv_cache -> output cache regions (key+value contiguous in both)
  const int n4 = (int)((2 * cache_elems) / 4);
  cache_copy_kernel<<<4096, 256, 0, stream>>>((const float4*)kvc, (float4*)kc, n4);

  // 2) scatter new tokens into the caches
  cache_scatter_kernel<<<BB * SS, 256, 0, stream>>>(k, v, bidx, boff, kc, vc);

  // 3) flash-attention prefill
  dim3 grid(SS / 128, HH, BB);
  attn_prefill_kernel<<<grid, 256, 0, stream>>>(q, k, v, out_attn);
}
